// AtomTransformer_57853209477445
// MI455X (gfx1250) — matrix operand order repaired and checked by reading, not compiled, not run on hardware
//
#include <hip/hip_runtime.h>

// ---------------------------------------------------------------------------
// AtomTransformer for MI455X (gfx1250): bf16 WMMA GEMMs + fused windowed attn.
// B=2 N=8192 C_A=128 C_S=384 C_Z=16 H=4 D=32 NQ=32 NK=128 NB=3 W=256 HID=256
// ---------------------------------------------------------------------------

typedef __attribute__((ext_vector_type(16))) __bf16 v16bf;
typedef __attribute__((ext_vector_type(8)))  float  v8f;

union FragB { v16bf v; unsigned u[8]; uint4 q[2]; };
union FragF { v8f  v; float    f[8]; };

__device__ __forceinline__ unsigned short f2bf(float f) {
    unsigned u = __float_as_uint(f);
    u += 0x7fffu + ((u >> 16) & 1u);           // round-to-nearest-even
    return (unsigned short)(u >> 16);
}
__device__ __forceinline__ unsigned pack2(float a, float b) {
    return (unsigned)f2bf(a) | ((unsigned)f2bf(b) << 16);
}
__device__ __forceinline__ float sigmoidf(float x) { return 1.0f / (1.0f + __expf(-x)); }

// A-fragment K offset for 16-bit WMMA (lane half h16, vgpr j): ISA 7.12.2
#define AMAP(j, h16) (((j) < 4) ? (8 * (h16) + 2 * (j)) : (16 + 8 * (h16) + 2 * ((j) - 4)))

// ------------------------- LN(s) -> bf16 (C_S = 384) -----------------------
__global__ __launch_bounds__(256) void sln_kernel(const float* __restrict__ s,
                                                  unsigned short* __restrict__ out) {
    const int wave = threadIdx.x >> 5, lane = threadIdx.x & 31;
    const size_t row = (size_t)blockIdx.x * 8 + wave;
    float4 x[3];
    float sum = 0.f, ssq = 0.f;
    for (int i = 0; i < 3; i++) {
        x[i] = *(const float4*)(s + row * 384 + i * 128 + lane * 4);
        sum += x[i].x + x[i].y + x[i].z + x[i].w;
        ssq += x[i].x * x[i].x + x[i].y * x[i].y + x[i].z * x[i].z + x[i].w * x[i].w;
    }
    for (int sh = 1; sh < 32; sh <<= 1) {
        sum += __shfl_xor(sum, sh, 32);
        ssq += __shfl_xor(ssq, sh, 32);
    }
    const float mean = sum * (1.f / 384.f);
    const float var = ssq * (1.f / 384.f) - mean * mean;
    const float rs = rsqrtf(var + 1e-5f);
    for (int i = 0; i < 3; i++) {
        uint2 r;
        r.x = pack2((x[i].x - mean) * rs, (x[i].y - mean) * rs);
        r.y = pack2((x[i].z - mean) * rs, (x[i].w - mean) * rs);
        *(uint2*)(out + row * 384 + i * 128 + lane * 4) = r;
    }
}

// ------------- LN(a) + two adaLN modulations -> bf16 (C_A = 128) -----------
__global__ __launch_bounds__(256) void ln_mod_kernel(
    const float* __restrict__ a, const float* __restrict__ sc1,
    const float* __restrict__ sh1, const float* __restrict__ sc2,
    const float* __restrict__ sh2, unsigned short* __restrict__ out1,
    unsigned short* __restrict__ out2) {
    const int wave = threadIdx.x >> 5, lane = threadIdx.x & 31;
    const size_t row = (size_t)blockIdx.x * 8 + wave;
    const size_t base = row * 128 + lane * 4;
    float4 x = *(const float4*)(a + base);
    float sum = x.x + x.y + x.z + x.w;
    float ssq = x.x * x.x + x.y * x.y + x.z * x.z + x.w * x.w;
    for (int sh = 1; sh < 32; sh <<= 1) {
        sum += __shfl_xor(sum, sh, 32);
        ssq += __shfl_xor(ssq, sh, 32);
    }
    const float mean = sum * (1.f / 128.f);
    const float var = ssq * (1.f / 128.f) - mean * mean;
    const float rs = rsqrtf(var + 1e-5f);
    float xn[4] = {(x.x - mean) * rs, (x.y - mean) * rs, (x.z - mean) * rs, (x.w - mean) * rs};
    float4 c1 = *(const float4*)(sc1 + base), h1 = *(const float4*)(sh1 + base);
    float4 c2 = *(const float4*)(sc2 + base), h2 = *(const float4*)(sh2 + base);
    uint2 r1, r2;
    r1.x = pack2(c1.x * xn[0] + h1.x, c1.y * xn[1] + h1.y);
    r1.y = pack2(c1.z * xn[2] + h1.z, c1.w * xn[3] + h1.w);
    r2.x = pack2(c2.x * xn[0] + h2.x, c2.y * xn[1] + h2.y);
    r2.y = pack2(c2.z * xn[2] + h2.z, c2.w * xn[3] + h2.w);
    *(uint2*)(out1 + base) = r1;
    *(uint2*)(out2 + base) = r2;
}

// --------- bias = (LN(p)*lnz_w + lnz_b) @ wb_pair -> [b,w,h,q,k] -----------
__global__ __launch_bounds__(256) void bias_kernel(
    const float* __restrict__ p, const float* __restrict__ lnzw,
    const float* __restrict__ lnzb, const float* __restrict__ wbp,
    float* __restrict__ biasbuf) {
    __shared__ float s_w[16], s_b[16], s_wb[64];
    if (threadIdx.x < 16) { s_w[threadIdx.x] = lnzw[threadIdx.x]; s_b[threadIdx.x] = lnzb[threadIdx.x]; }
    if (threadIdx.x < 64) s_wb[threadIdx.x] = wbp[threadIdx.x];
    __syncthreads();
    const size_t idx = (size_t)blockIdx.x * 256 + threadIdx.x;   // b*2^20 + w*4096 + q*128 + k
    const float* pp = p + idx * 16;
    float v[16], sum = 0.f, ssq = 0.f;
    for (int c = 0; c < 16; c++) { v[c] = pp[c]; sum += v[c]; ssq += v[c] * v[c]; }
    const float mean = sum * (1.f / 16.f);
    const float var = ssq * (1.f / 16.f) - mean * mean;
    const float rs = rsqrtf(var + 1e-5f);
    float acc[4] = {0.f, 0.f, 0.f, 0.f};
    for (int c = 0; c < 16; c++) {
        float y = (v[c] - mean) * rs * s_w[c] + s_b[c];
        for (int hh = 0; hh < 4; hh++) acc[hh] += y * s_wb[c * 4 + hh];
    }
    const int b = (int)(idx >> 20);
    const int w = (int)((idx >> 12) & 255);
    const int qk = (int)(idx & 4095);
    const size_t base = (((size_t)(b * 256 + w) * 4) << 12) + qk;
    for (int hh = 0; hh < 4; hh++) biasbuf[base + ((size_t)hh << 12)] = acc[hh];
}

// --------- convert W[K][N] f32 -> Wt[N][K] bf16 (transposed weights) -------
__global__ __launch_bounds__(256) void convT_kernel(const float* __restrict__ src,
                                                    unsigned short* __restrict__ dst,
                                                    int K, int N) {
    const int i = blockIdx.x * 256 + threadIdx.x;
    if (i >= K * N) return;
    const int n = i / K, k = i % K;
    dst[i] = f2bf(src[(size_t)k * N + n]);
}

// ------------------ generic bf16 WMMA GEMM with fused epilogues ------------
// A: MxK bf16 row-major.  Wt/Wt2: NxK bf16 ("B transposed").  Out width = N.
// MODE 0: (+bias) -> bf16      1: -> f32            2: sigmoid(+bias) -> f32
//      3: x*gate -> f32        4: outF += x*gate    5: silu(x1)*x2 -> bf16
// Double-buffered LDS weight staging; padded rows (40 hw = 80B) keep b128
// fragment loads 16B-aligned and bank-conflict free.
template <int MODE>
__global__ __launch_bounds__(256) void gemm_bf16_kernel(
    const unsigned short* __restrict__ A, const unsigned short* __restrict__ Wt,
    const unsigned short* __restrict__ Wt2, const float* __restrict__ bias,
    const float* __restrict__ gate, float* __restrict__ outF,
    unsigned short* __restrict__ outB, int M, int K, int N) {
    constexpr int NMAT = (MODE == 5) ? 2 : 1;
    __shared__ unsigned short ldsB[2][NMAT][128][40];
    const int wave = threadIdx.x >> 5, lane = threadIdx.x & 31;
    const int m = lane & 15, h16 = lane >> 4;
    const int m0 = blockIdx.x * 128, n0 = blockIdx.y * 128;

    FragF acc[8], acc2[8];
    for (int t = 0; t < 8; t++)
        for (int e = 0; e < 8; e++) { acc[t].f[e] = 0.f; acc2[t].f[e] = 0.f; }

    const int r_st = threadIdx.x >> 1, hh_st = threadIdx.x & 1;
    auto stage = [&](int k0, int ph) {
        const uint4* src = (const uint4*)(Wt + (size_t)(n0 + r_st) * K + k0 + hh_st * 16);
        uint4* dst = (uint4*)&ldsB[ph][0][r_st][hh_st * 16];
        dst[0] = src[0];
        dst[1] = src[1];
        if (MODE == 5) {
            const uint4* src2 = (const uint4*)(Wt2 + (size_t)(n0 + r_st) * K + k0 + hh_st * 16);
            uint4* dst2 = (uint4*)&ldsB[ph][1][r_st][hh_st * 16];
            dst2[0] = src2[0];
            dst2[1] = src2[1];
        }
    };

    const size_t arow = (size_t)m0 + wave * 16 + m;
    const int nk = K >> 5;
    stage(0, 0);
    int ph = 0;
    for (int ks = 0; ks < nk; ks++) {
        __syncthreads();
        if (ks + 1 < nk) stage((ks + 1) << 5, ph ^ 1);
        const int k0 = ks << 5;
        FragB a;
        a.q[0] = *(const uint4*)&A[arow * K + k0 + 8 * h16];
        a.q[1] = *(const uint4*)&A[arow * K + k0 + 16 + 8 * h16];
        for (int nt = 0; nt < 8; nt++) {
            FragB b;
            b.q[0] = *(const uint4*)&ldsB[ph][0][nt * 16 + m][8 * h16];
            b.q[1] = *(const uint4*)&ldsB[ph][0][nt * 16 + m][16 + 8 * h16];
            acc[nt].v = __builtin_amdgcn_wmma_f32_16x16x32_bf16(
                false, a.v, false, b.v, (short)0, acc[nt].v, false, false);
            if (MODE == 5) {
                FragB b2;
                b2.q[0] = *(const uint4*)&ldsB[ph][1][nt * 16 + m][8 * h16];
                b2.q[1] = *(const uint4*)&ldsB[ph][1][nt * 16 + m][16 + 8 * h16];
                acc2[nt].v = __builtin_amdgcn_wmma_f32_16x16x32_bf16(
                    false, a.v, false, b2.v, (short)0, acc2[nt].v, false, false);
            }
        }
        ph ^= 1;
    }
    // epilogue (C/D layout: row = 8*h16 + j, col = lane&15)
    for (int nt = 0; nt < 8; nt++) {
        for (int j = 0; j < 8; j++) {
            const size_t r = (size_t)m0 + wave * 16 + 8 * h16 + j;
            const size_t c = (size_t)n0 + nt * 16 + m;
            const size_t o = r * (size_t)N + c;
            float x = acc[nt].f[j];
            if (MODE == 0) {
                if (bias) x += bias[c];
                outB[o] = f2bf(x);
            } else if (MODE == 1) {
                outF[o] = x;
            } else if (MODE == 2) {
                if (bias) x += bias[c];
                outF[o] = sigmoidf(x);
            } else if (MODE == 3) {
                outF[o] = x * gate[o];
            } else if (MODE == 4) {
                outF[o] += x * gate[o];
            } else {   // 5: silu(x)*x2 -> bf16
                outB[o] = f2bf((x * sigmoidf(x)) * acc2[nt].f[j]);
            }
        }
    }
}

// ----------------- fused windowed attention (1 wave = 1 head) --------------
// grid = B*W blocks, 128 threads (4 waves). q/k/v bf16 [B*N][128].
// Logits + softmax are done in two independent 16-query passes so only 8
// accumulator tiles (64 VGPRs) are live at once -> no scratch spills.
__global__ __launch_bounds__(128) void attn_kernel(
    const unsigned short* __restrict__ qb, const unsigned short* __restrict__ kb,
    const unsigned short* __restrict__ vb, const float* __restrict__ gbuf,
    const float* __restrict__ biasbuf, unsigned short* __restrict__ og) {
    __shared__ unsigned short ap[4][32][136];    // per-head probs, bf16, 16B-aligned rows
    const int h = threadIdx.x >> 5;
    const int lane = threadIdx.x & 31;
    const int m = lane & 15, h16 = lane >> 4;
    const int bw = blockIdx.x;
    const int b = bw >> 8, w = bw & 255;
    const size_t rowq0 = (size_t)b * 8192 + (size_t)w * 32;
    const float inv_sqrt_d = 0.17677669529663687f;   // 1/sqrt(32)
    const float* bb = biasbuf + (((size_t)bw * 4 + h) << 12);

    // ---- pass over the two 16-query M-tiles: logits -> softmax -> LDS ----
    for (int mt = 0; mt < 2; mt++) {
        // Q fragment (A-matrix, K = D = 32: exactly one bf16 WMMA k-step)
        FragB aq;
        {
            const size_t qoff = (rowq0 + mt * 16 + m) * 128 + h * 32;
            aq.q[0] = *(const uint4*)&qb[qoff + 8 * h16];
            aq.q[1] = *(const uint4*)&qb[qoff + 16 + 8 * h16];
        }
        FragF lt[8];
        for (int nt = 0; nt < 8; nt++) {
            FragB bk;
            {
                int kk = nt * 16 + m;
                int kidx = w * 32 - 48 + kk;
                kidx = kidx < 0 ? 0 : (kidx > 8191 ? 8191 : kidx);
                const size_t koff = ((size_t)b * 8192 + kidx) * 128 + h * 32 + 8 * h16;
                bk.q[0] = *(const uint4*)&kb[koff];
                bk.q[1] = *(const uint4*)&kb[koff + 16];
            }
            FragF z;
            for (int e = 0; e < 8; e++) z.f[e] = 0.f;
            lt[nt].v = __builtin_amdgcn_wmma_f32_16x16x32_bf16(
                false, aq.v, false, bk.v, (short)0, z.v, false, false);
        }
        // scale + pair bias + window mask
        for (int nt = 0; nt < 8; nt++)
            for (int j = 0; j < 8; j++) {
                const int qr = mt * 16 + 8 * h16 + j;
                const int kk = nt * 16 + m;
                const int kidx = w * 32 - 48 + kk;
                float v = lt[nt].f[j] * inv_sqrt_d + bb[qr * 128 + kk];
                if (kidx < 0 || kidx >= 8192) v = -1e9f;
                lt[nt].f[j] = v;
            }
        // softmax per query row (row lives in one 16-lane half across 8 n-tiles)
        for (int j = 0; j < 8; j++) {
            float mx = -3.4e38f;
            for (int nt = 0; nt < 8; nt++) mx = fmaxf(mx, lt[nt].f[j]);
            for (int s = 1; s < 16; s <<= 1) mx = fmaxf(mx, __shfl_xor(mx, s, 32));
            float sum = 0.f;
            for (int nt = 0; nt < 8; nt++) {
                float e = __expf(lt[nt].f[j] - mx);
                lt[nt].f[j] = e;
                sum += e;
            }
            for (int s = 1; s < 16; s <<= 1) sum += __shfl_xor(sum, s, 32);
            const float inv = 1.0f / sum;
            for (int nt = 0; nt < 8; nt++) lt[nt].f[j] *= inv;
        }
        // stage probs as bf16 [q][k] in LDS to re-fragment for attn @ V
        for (int nt = 0; nt < 8; nt++)
            for (int j = 0; j < 8; j++)
                ap[h][mt * 16 + 8 * h16 + j][nt * 16 + m] = f2bf(lt[nt].f[j]);
    }
    __syncthreads();

    // O = attn @ V  (M=32 queries, N=32 d, K=128 keys -> 4 k-steps)
    FragF accO[2][2];
    for (int mt = 0; mt < 2; mt++)
        for (int dt = 0; dt < 2; dt++)
            for (int e = 0; e < 8; e++) accO[mt][dt].f[e] = 0.f;
    for (int ks = 0; ks < 4; ks++) {
        FragB a2[2];
        for (int mt = 0; mt < 2; mt++) {
            a2[mt].q[0] = *(const uint4*)&ap[h][mt * 16 + m][ks * 32 + 8 * h16];
            a2[mt].q[1] = *(const uint4*)&ap[h][mt * 16 + m][ks * 32 + 16 + 8 * h16];
        }
        for (int dt = 0; dt < 2; dt++) {
            FragB bv;
            for (int j = 0; j < 8; j++) {
                const int k0 = ks * 32 + AMAP(j, h16);
                int ki0 = w * 32 - 48 + k0, ki1 = ki0 + 1;
                ki0 = ki0 < 0 ? 0 : (ki0 > 8191 ? 8191 : ki0);
                ki1 = ki1 < 0 ? 0 : (ki1 > 8191 ? 8191 : ki1);
                const unsigned lo = vb[((size_t)b * 8192 + ki0) * 128 + h * 32 + dt * 16 + m];
                const unsigned hi = vb[((size_t)b * 8192 + ki1) * 128 + h * 32 + dt * 16 + m];
                bv.u[j] = lo | (hi << 16);
            }
            for (int mt = 0; mt < 2; mt++)
                accO[mt][dt].v = __builtin_amdgcn_wmma_f32_16x16x32_bf16(
                    false, a2[mt].v, false, bv.v, (short)0, accO[mt][dt].v, false, false);
        }
    }
    // gate and emit bf16 for the wo projection
    for (int mt = 0; mt < 2; mt++)
        for (int dt = 0; dt < 2; dt++)
            for (int j = 0; j < 8; j++) {
                const int qr = mt * 16 + 8 * h16 + j;
                const size_t grow = rowq0 + qr;
                const int c = h * 32 + dt * 16 + m;
                const float val = accO[mt][dt].f[j] * gbuf[grow * 128 + c];
                og[grow * 128 + c] = f2bf(val);
            }
}

// ---------------------------------------------------------------------------
extern "C" void kernel_launch(void* const* d_in, const int* in_sizes, int n_in,
                              void* d_out, int out_size, void* d_ws, size_t ws_size,
                              hipStream_t stream) {
    (void)in_sizes; (void)n_in; (void)out_size;
    const float* q_in = (const float*)d_in[0];
    const float* s_in = (const float*)d_in[1];
    const float* p_in = (const float*)d_in[2];
    const float* adaln_scale_w = (const float*)d_in[3];
    const float* adaln_scale_b = (const float*)d_in[4];
    const float* adaln_shift_w = (const float*)d_in[5];
    const float* wq = (const float*)d_in[6];
    const float* bq = (const float*)d_in[7];
    const float* wk = (const float*)d_in[8];
    const float* wv = (const float*)d_in[9];
    const float* lnz_w = (const float*)d_in[10];
    const float* lnz_b = (const float*)d_in[11];
    const float* wb_pair = (const float*)d_in[12];
    const float* wg = (const float*)d_in[13];
    const float* wo = (const float*)d_in[14];
    const float* sgate_w = (const float*)d_in[15];
    const float* sgate_b = (const float*)d_in[16];
    const float* t_scale_w = (const float*)d_in[17];
    const float* t_scale_b = (const float*)d_in[18];
    const float* t_shift_w = (const float*)d_in[19];
    const float* t_wa = (const float*)d_in[20];
    const float* t_wb = (const float*)d_in[21];
    const float* t_wo = (const float*)d_in[22];
    const float* t_sgate_w = (const float*)d_in[23];
    const float* t_sgate_b = (const float*)d_in[24];

    constexpr int CA = 128, CS = 384, CZ = 16, Hh = 4, HD = 128, HID = 256;
    constexpr size_t Mr = 16384;   // B*N

    char* ws = (char*)d_ws;
    size_t off = 0;
    auto alloc = [&](size_t bytes) -> void* {
        void* p = ws + off;
        off = (off + bytes + 255) & ~(size_t)255;
        return p;
    };
    unsigned short* slnB = (unsigned short*)alloc(Mr * CS * 2);
    float* modScale = (float*)alloc(Mr * CA * 4);
    float* modShift = (float*)alloc(Mr * CA * 4);
    float* sGate = (float*)alloc(Mr * CA * 4);
    float* tScale = (float*)alloc(Mr * CA * 4);
    float* tShift = (float*)alloc(Mr * CA * 4);
    float* tGate = (float*)alloc(Mr * CA * 4);
    unsigned short* alnB = (unsigned short*)alloc(Mr * CA * 2);
    unsigned short* tlnB = (unsigned short*)alloc(Mr * CA * 2);
    unsigned short* qB = (unsigned short*)alloc(Mr * CA * 2);
    unsigned short* kB = (unsigned short*)alloc(Mr * CA * 2);
    unsigned short* vB = (unsigned short*)alloc(Mr * CA * 2);
    float* gF = (float*)alloc(Mr * CA * 4);
    unsigned short* ogB = (unsigned short*)alloc(Mr * CA * 2);
    unsigned short* hidB = (unsigned short*)alloc(Mr * HID * 2);
    float* biasBuf = (float*)alloc((size_t)2 * 256 * 4 * 32 * 128 * 4);
    float* aBuf0 = (float*)alloc(Mr * CA * 4);
    float* aBuf1 = (float*)alloc(Mr * CA * 4);
    unsigned short* wT = (unsigned short*)alloc(475136ull * 2);
    if (off > ws_size) return;   // workspace too small (deterministic no-op)

    sln_kernel<<<2048, 256, 0, stream>>>(s_in, slnB);

    const float* aCur = q_in;
    float* outs[3] = {aBuf0, aBuf1, (float*)d_out};
    const dim3 g128(128, 1), g256(128, 2);

    for (int i = 0; i < 3; i++) {
        // ---- pre-transpose+convert this block's weights to bf16 [N][K] ----
        auto convT = [&](const float* src, unsigned short* dst, int K, int N) {
            convT_kernel<<<(K * N + 255) / 256, 256, 0, stream>>>(src, dst, K, N);
        };
        unsigned short* adsW = wT;
        unsigned short* adhW = wT + 49152;
        unsigned short* sgW = wT + 2 * 49152;
        unsigned short* tscW = wT + 3 * 49152;
        unsigned short* tshW = wT + 4 * 49152;
        unsigned short* tsgW = wT + 5 * 49152;
        unsigned short* wqT = wT + 6 * 49152;
        unsigned short* wkT = wqT + 16384;
        unsigned short* wvT = wkT + 16384;
        unsigned short* wgT = wvT + 16384;
        unsigned short* woT = wgT + 16384;
        unsigned short* waT = woT + 16384;
        unsigned short* wbT = waT + 32768;
        unsigned short* wotT = wbT + 32768;
        convT(adaln_scale_w + (size_t)i * CS * CA, adsW, CS, CA);
        convT(adaln_shift_w + (size_t)i * CS * CA, adhW, CS, CA);
        convT(sgate_w + (size_t)i * CS * CA, sgW, CS, CA);
        convT(t_scale_w + (size_t)i * CS * CA, tscW, CS, CA);
        convT(t_shift_w + (size_t)i * CS * CA, tshW, CS, CA);
        convT(t_sgate_w + (size_t)i * CS * CA, tsgW, CS, CA);
        convT(wq + (size_t)i * CA * HD, wqT, CA, HD);
        convT(wk + (size_t)i * CA * HD, wkT, CA, HD);
        convT(wv + (size_t)i * CA * HD, wvT, CA, HD);
        convT(wg + (size_t)i * CA * HD, wgT, CA, HD);
        convT(wo + (size_t)i * HD * CA, woT, HD, CA);
        convT(t_wa + (size_t)i * CA * HID, waT, CA, HID);
        convT(t_wb + (size_t)i * CA * HID, wbT, CA, HID);
        convT(t_wo + (size_t)i * HID * CA, wotT, HID, CA);

        // ---- six s_ln projections (K=384, N=128) ----
        gemm_bf16_kernel<2><<<g128, 256, 0, stream>>>(slnB, adsW, nullptr,
            adaln_scale_b + i * CA, nullptr, modScale, nullptr, (int)Mr, CS, CA);
        gemm_bf16_kernel<1><<<g128, 256, 0, stream>>>(slnB, adhW, nullptr,
            nullptr, nullptr, modShift, nullptr, (int)Mr, CS, CA);
        gemm_bf16_kernel<2><<<g128, 256, 0, stream>>>(slnB, sgW, nullptr,
            sgate_b + i * CA, nullptr, sGate, nullptr, (int)Mr, CS, CA);
        gemm_bf16_kernel<2><<<g128, 256, 0, stream>>>(slnB, tscW, nullptr,
            t_scale_b + i * CA, nullptr, tScale, nullptr, (int)Mr, CS, CA);
        gemm_bf16_kernel<1><<<g128, 256, 0, stream>>>(slnB, tshW, nullptr,
            nullptr, nullptr, tShift, nullptr, (int)Mr, CS, CA);
        gemm_bf16_kernel<2><<<g128, 256, 0, stream>>>(slnB, tsgW, nullptr,
            t_sgate_b + i * CA, nullptr, tGate, nullptr, (int)Mr, CS, CA);

        // ---- pair bias from p (single pass over p per block) ----
        bias_kernel<<<8192, 256, 0, stream>>>(p_in, lnz_w + i * CZ, lnz_b + i * CZ,
                                              wb_pair + (size_t)i * CZ * Hh, biasBuf);

        // ---- LN(a) + both modulations ----
        ln_mod_kernel<<<2048, 256, 0, stream>>>(aCur, modScale, modShift, tScale,
                                                tShift, alnB, tlnB);

        // ---- q, k, v, g projections (K=128, N=128) ----
        gemm_bf16_kernel<0><<<g128, 256, 0, stream>>>(alnB, wqT, nullptr, bq + i * HD,
            nullptr, nullptr, qB, (int)Mr, CA, HD);
        gemm_bf16_kernel<0><<<g128, 256, 0, stream>>>(alnB, wkT, nullptr, nullptr,
            nullptr, nullptr, kB, (int)Mr, CA, HD);
        gemm_bf16_kernel<0><<<g128, 256, 0, stream>>>(alnB, wvT, nullptr, nullptr,
            nullptr, nullptr, vB, (int)Mr, CA, HD);
        gemm_bf16_kernel<2><<<g128, 256, 0, stream>>>(alnB, wgT, nullptr, nullptr,
            nullptr, gF, nullptr, (int)Mr, CA, HD);

        // ---- fused windowed attention ----
        attn_kernel<<<512, 128, 0, stream>>>(qB, kB, vB, gF, biasBuf, ogB);

        // ---- attn_out = (o @ wo) * sgate ----
        gemm_bf16_kernel<3><<<g128, 256, 0, stream>>>(ogB, woT, nullptr, nullptr,
            sGate, outs[i], nullptr, (int)Mr, HD, CA);

        // ---- hidden = silu(t_ln@t_wa) * (t_ln@t_wb) (K=128, N=256) ----
        gemm_bf16_kernel<5><<<g256, 256, 0, stream>>>(tlnB, waT, wbT, nullptr,
            nullptr, nullptr, hidB, (int)Mr, CA, HID);

        // ---- a = attn_out + (hidden @ t_wo) * t_sgate ----
        gemm_bf16_kernel<4><<<g128, 256, 0, stream>>>(hidB, wotT, nullptr, nullptr,
            tGate, outs[i], nullptr, (int)Mr, HID, CA);

        aCur = outs[i];
    }
}
